// Model_18305150616261
// MI455X (gfx1250) — matrix operand order repaired and checked by reading, not compiled, not run on hardware
//
#include <hip/hip_runtime.h>
#include <hip/hip_bf16.h>

typedef __attribute__((ext_vector_type(16))) _Float16 v16h;
typedef __attribute__((ext_vector_type(8)))  _Float16 v8h;
typedef __attribute__((ext_vector_type(8)))  float    v8f;

#define B_   64
#define T_   512
#define WIN  40
#define T2   510          // T - 2 (VALID 3x3)
#define W2   38           // WIN - 2
#define CF   8
#define GH   32
#define GIN  304          // W2*CF
#define KPAD 320          // GIN padded to 10 x 32 for WMMA K
#define G3   96           // 3*GH
#define ATTN 32

// ---- workspace layout (bytes) ----
#define OFF_SEQ  ((size_t)0)                         // f16 [T2][B][KPAD]
#define SZ_SEQ   ((size_t)T2*B_*KPAD*2)              // 20,889,600
#define OFF_WIH  (OFF_SEQ + SZ_SEQ)                  // f16 [96][KPAD]
#define SZ_WIH   ((size_t)G3*KPAD*2)
#define OFF_WHH  (OFF_WIH + SZ_WIH)                  // f16 [96][32]
#define SZ_WHH   ((size_t)G3*GH*2)
#define OFF_XP   (OFF_WHH + SZ_WHH)                  // f32 [T2*B][96]
#define SZ_XP    ((size_t)T2*B_*G3*4)
#define OFF_ENC  (OFF_XP + SZ_XP)                    // f32 [T2][B][GH]
#define SZ_ENC   ((size_t)T2*B_*GH*4)
#define OFF_A    (OFF_ENC + SZ_ENC)                  // f32 [B][T2][GH] attention logits
#define SZ_A     ((size_t)B_*T2*GH*4)

// ============================================================================
// Kernel 1: fused deformable-conv front end.
// One thread per (b, i, j) output point: 18 offset channels from a 3x3 VALID
// conv, 9 bilinear samples, 8 conv filters -> writes seq in (t, b, j*CF+c)
// layout directly as f16 (16B vector store), plus zero K-padding.
// ============================================================================
__global__ void k_frontend(const float* __restrict__ x,
                           const float* __restrict__ p_w, const float* __restrict__ p_b,
                           const float* __restrict__ dcn_w, const float* __restrict__ dcn_b,
                           _Float16* __restrict__ seq)
{
    int tid = blockIdx.x * blockDim.x + threadIdx.x;
    int b   = tid / (T2 * W2);
    int rem = tid - b * (T2 * W2);
    int i   = rem / W2;
    int j   = rem - i * W2;

    const float* xb = x + (size_t)b * T_ * WIN;

    float patch[9];
#pragma unroll
    for (int u = 0; u < 3; ++u)
#pragma unroll
        for (int v = 0; v < 3; ++v)
            patch[u * 3 + v] = xb[(i + u) * WIN + (j + v)];

    float off[18];
#pragma unroll
    for (int c = 0; c < 18; ++c) {
        float s = p_b[c];
#pragma unroll
        for (int k = 0; k < 9; ++k) s += p_w[c * 9 + k] * patch[k];
        off[c] = s;
    }

    float samp[9];
#pragma unroll
    for (int n = 0; n < 9; ++n) {
        // p0 = (i+1, j+1), pn = (n/3 - 1, n%3 - 1)
        float py = off[n]     + (float)(n / 3) + (float)i;
        float px = off[9 + n] + (float)(n % 3) + (float)j;
        py = fminf(fmaxf(py, 0.0f), (float)(T_ - 1));
        px = fminf(fmaxf(px, 0.0f), (float)(WIN - 1));
        int y0 = (int)floorf(py);
        int x0 = (int)floorf(px);
        int y1 = min(y0 + 1, T_ - 1);
        int x1 = min(x0 + 1, WIN - 1);
        float wy = py - (float)y0;
        float wx = px - (float)x0;
        float g00 = xb[y0 * WIN + x0];
        float g01 = xb[y0 * WIN + x1];
        float g10 = xb[y1 * WIN + x0];
        float g11 = xb[y1 * WIN + x1];
        samp[n] = g00 * (1.0f - wy) * (1.0f - wx) + g01 * (1.0f - wy) * wx
                + g10 * wy * (1.0f - wx)          + g11 * wy * wx;
    }

    v8h o16;
#pragma unroll
    for (int c = 0; c < CF; ++c) {
        float s = dcn_b[c];
#pragma unroll
        for (int n = 0; n < 9; ++n) s += dcn_w[c * 9 + n] * samp[n];
        o16[c] = (_Float16)s;
    }

    _Float16* row = seq + ((size_t)i * B_ + b) * KPAD;
    *(v8h*)(row + j * CF) = o16;          // 16B aligned vector store
    if (j < KPAD - GIN)                   // zero the K padding (16 halves/row)
        row[GIN + j] = (_Float16)0.0f;
}

// ============================================================================
// Kernel 2: weight conversion f32 -> f16 (w_ih padded K 304->320, w_hh).
// ============================================================================
__global__ void k_prep_w(const float* __restrict__ w_ih, const float* __restrict__ w_hh,
                         _Float16* __restrict__ wih16, _Float16* __restrict__ whh16)
{
    int idx = blockIdx.x * blockDim.x + threadIdx.x;   // 33792 threads exactly
    if (idx < G3 * KPAD) {
        int n = idx / KPAD, k = idx - n * KPAD;
        wih16[idx] = (_Float16)(k < GIN ? w_ih[n * GIN + k] : 0.0f);
    } else {
        int r = idx - G3 * KPAD;                        // < 96*32
        whh16[r] = (_Float16)w_hh[r];
    }
}

// ============================================================================
// Kernel 3: xp = seq(32640 x 320) @ w_ih^T(320 x 96) via v_wmma_f32_16x16x32_f16.
// One wave per 16-row M tile, six v8f accumulators span all of N=96.
// Fragment layouts (this code's contiguous k order, same in A and B):
//   A (16x32 f16): lane(hi,m): 16 contiguous halves of row m at k0+16*hi
//   B (32x16 f16): lane(khalf,n): 16 contiguous halves of column n at k0+16*khalf
//   C (16x16 f32): VGPR r -> row r + 8*hi, col = lane&15
// ============================================================================
__global__ __launch_bounds__(256) void k_xproj(const _Float16* __restrict__ seq,
                                               const _Float16* __restrict__ wih16,
                                               float* __restrict__ xp)
{
    int gtid = blockIdx.x * 256 + threadIdx.x;
    int wave = gtid >> 5;          // 0..2039 (M tiles)
    int lane = threadIdx.x & 31;
    int nl   = lane & 15;
    int hi   = lane >> 4;
    int m0   = wave << 4;

    const _Float16* arow = seq + (size_t)(m0 + nl) * KPAD;

    v8f acc[6] = {};
#pragma unroll
    for (int k0 = 0; k0 < KPAD; k0 += 32) {
        __builtin_prefetch(arow + k0 + 64, 0, 1);   // global_prefetch next A chunk
        v8h alo = *(const v8h*)(arow + k0 + hi * 16);
        v8h ahi = *(const v8h*)(arow + k0 + 8 + hi * 16);
        v16h a  = __builtin_shufflevector(alo, ahi,
                    0,1,2,3,4,5,6,7,8,9,10,11,12,13,14,15);
#pragma unroll
        for (int n = 0; n < 6; ++n) {
            const _Float16* bp = wih16 + (size_t)(n * 16 + nl) * KPAD + k0 + hi * 16;
            v16h bfrag = *(const v16h*)bp;
            acc[n] = __builtin_amdgcn_wmma_f32_16x16x32_f16(
                         false, a, false, bfrag, (short)0, acc[n], false, false);
        }
    }

#pragma unroll
    for (int n = 0; n < 6; ++n)
#pragma unroll
        for (int r = 0; r < 8; ++r)
            xp[(size_t)(m0 + hi * 8 + r) * G3 + n * 16 + nl] = acc[n][r];
}

// ============================================================================
// Kernel 4: GRU scan. 4 blocks x 16 batches, 6 waves/block (one N tile each).
// h state lives in LDS (f32 + f16 mirror); w_hh^T fragments preloaded into
// registers once and reused across all 510 steps; one WMMA per wave per step.
// ============================================================================
__global__ __launch_bounds__(192) void k_gru(const float* __restrict__ xp,
                                             const _Float16* __restrict__ whh16,
                                             const float* __restrict__ b_ih,
                                             const float* __restrict__ b_hh,
                                             float* __restrict__ enc)
{
    __shared__ __align__(16) float    sh_h[16 * GH];     // f32 state, 16 batches
    __shared__ __align__(16) _Float16 sh_h16[16 * GH];   // f16 mirror for WMMA A
    __shared__ __align__(16) float    sh_hp[16 * G3];    // h @ w_hh^T

    const int tid   = threadIdx.x;
    const int wave  = tid >> 5;         // 0..5 = N tile
    const int lane  = tid & 31;
    const int nl    = lane & 15;
    const int hi    = lane >> 4;
    const int bbase = blockIdx.x * 16;

    // Preload this wave's B fragment of w_hh^T (K=32, one WMMA step) — lives in
    // 8 VGPRs for the entire scan.
    v16h bfrag = *(const v16h*)(whh16 + (size_t)(wave * 16 + nl) * GH + hi * 16);

    for (int idx = tid; idx < 16 * GH; idx += 192) {
        sh_h[idx]   = 0.0f;
        sh_h16[idx] = (_Float16)0.0f;
    }
    __syncthreads();

    for (int t = 0; t < T2; ++t) {
        // A fragment: rows m = nl (local batch), K contiguous at 16*hi
        v8h alo = *(const v8h*)(&sh_h16[nl * GH + hi * 16]);
        v8h ahi = *(const v8h*)(&sh_h16[nl * GH + 8 + hi * 16]);
        v16h a  = __builtin_shufflevector(alo, ahi,
                    0,1,2,3,4,5,6,7,8,9,10,11,12,13,14,15);
        v8f c = {};
        c = __builtin_amdgcn_wmma_f32_16x16x32_f16(
                false, a, false, bfrag, (short)0, c, false, false);
#pragma unroll
        for (int r = 0; r < 8; ++r)
            sh_hp[(hi * 8 + r) * G3 + wave * 16 + nl] = c[r];
        __syncthreads();

        const float* xpt = xp + (size_t)t * B_ * G3;
        for (int idx = tid; idx < 16 * GH; idx += 192) {
            int bloc  = idx >> 5;
            int g     = idx & 31;
            int bglob = bbase + bloc;
            float xr = xpt[bglob * G3 + g]          + b_ih[g];
            float xz = xpt[bglob * G3 + GH + g]     + b_ih[GH + g];
            float xn = xpt[bglob * G3 + 2 * GH + g] + b_ih[2 * GH + g];
            float hr = sh_hp[bloc * G3 + g]          + b_hh[g];
            float hz = sh_hp[bloc * G3 + GH + g]     + b_hh[GH + g];
            float hn = sh_hp[bloc * G3 + 2 * GH + g] + b_hh[2 * GH + g];
            float rg = 1.0f / (1.0f + expf(-(xr + hr)));
            float zg = 1.0f / (1.0f + expf(-(xz + hz)));
            float ng = tanhf(xn + rg * hn);
            float hprev = sh_h[idx];
            float hnew  = (1.0f - zg) * ng + zg * hprev;
            sh_h[idx]   = hnew;
            sh_h16[idx] = (_Float16)hnew;
            enc[((size_t)t * B_ + bglob) * GH + g] = hnew;
        }
        __syncthreads();
    }
}

// ============================================================================
// Kernel 5: attention pooling + classifier, one block per batch.
// ============================================================================
__global__ __launch_bounds__(256) void k_attn(const float* __restrict__ enc,
                                              const float* __restrict__ w1, const float* __restrict__ b1,
                                              const float* __restrict__ w2, const float* __restrict__ b2,
                                              const float* __restrict__ w3, const float* __restrict__ b3,
                                              float* __restrict__ aBuf,
                                              float* __restrict__ out)
{
    const int b   = blockIdx.x;
    const int tid = threadIdx.x;
    __shared__ float pooledS[GH];

    for (int t = tid; t < T2; t += 256) {
        float e[GH];
#pragma unroll
        for (int g = 0; g < GH; ++g) e[g] = enc[((size_t)t * B_ + b) * GH + g];
        float z[ATTN];
#pragma unroll
        for (int o = 0; o < ATTN; ++o) {
            float s = b1[o];
#pragma unroll
            for (int g = 0; g < GH; ++g) s += w1[o * GH + g] * e[g];
            z[o] = tanhf(s);
        }
#pragma unroll
        for (int o = 0; o < GH; ++o) {
            float s = b2[o];
#pragma unroll
            for (int j = 0; j < ATTN; ++j) s += w2[o * ATTN + j] * z[j];
            aBuf[((size_t)b * T2 + t) * GH + o] = s;
        }
    }
    __syncthreads();

    if (tid < GH) {
        int g = tid;
        float m = -1e30f;
        for (int t = 0; t < T2; ++t)
            m = fmaxf(m, aBuf[((size_t)b * T2 + t) * GH + g]);
        float s = 0.0f, p = 0.0f;
        for (int t = 0; t < T2; ++t) {
            float ev = expf(aBuf[((size_t)b * T2 + t) * GH + g] - m);
            s += ev;
            p += ev * enc[((size_t)t * B_ + b) * GH + g];
        }
        pooledS[g] = p / s;
    }
    __syncthreads();

    if (tid < 2) {
        float s = b3[tid];
#pragma unroll
        for (int g = 0; g < GH; ++g) s += w3[tid * GH + g] * pooledS[g];
        out[b * 2 + tid] = s;
    }
}

// ============================================================================
extern "C" void kernel_launch(void* const* d_in, const int* in_sizes, int n_in,
                              void* d_out, int out_size, void* d_ws, size_t ws_size,
                              hipStream_t stream)
{
    (void)in_sizes; (void)n_in; (void)out_size; (void)ws_size;
    const float* x     = (const float*)d_in[0];
    const float* p_w   = (const float*)d_in[1];
    const float* p_b   = (const float*)d_in[2];
    const float* dcn_w = (const float*)d_in[3];
    const float* dcn_b = (const float*)d_in[4];
    const float* w_ih  = (const float*)d_in[5];
    const float* w_hh  = (const float*)d_in[6];
    const float* b_ih  = (const float*)d_in[7];
    const float* b_hh  = (const float*)d_in[8];
    const float* w1    = (const float*)d_in[9];
    const float* b1    = (const float*)d_in[10];
    const float* w2    = (const float*)d_in[11];
    const float* b2    = (const float*)d_in[12];
    const float* w3    = (const float*)d_in[13];
    const float* b3    = (const float*)d_in[14];

    char* ws = (char*)d_ws;
    _Float16* seq16 = (_Float16*)(ws + OFF_SEQ);
    _Float16* wih16 = (_Float16*)(ws + OFF_WIH);
    _Float16* whh16 = (_Float16*)(ws + OFF_WHH);
    float*    xp    = (float*)(ws + OFF_XP);
    float*    enc   = (float*)(ws + OFF_ENC);
    float*    aBuf  = (float*)(ws + OFF_A);
    float*    out   = (float*)d_out;

    k_prep_w  <<<(G3 * KPAD + G3 * GH) / 256, 256, 0, stream>>>(w_ih, w_hh, wih16, whh16);
    k_frontend<<<(B_ * T2 * W2) / 256,        256, 0, stream>>>(x, p_w, p_b, dcn_w, dcn_b, seq16);
    k_xproj   <<<255,                         256, 0, stream>>>(seq16, wih16, xp);
    k_gru     <<<4,                           192, 0, stream>>>(xp, whh16, b_ih, b_hh, enc);
    k_attn    <<<B_,                          256, 0, stream>>>(enc, w1, b1, w2, b2, w3, b3, aBuf, out);
}
